// GCN_31593779429620
// MI455X (gfx1250) — matrix operand order repaired and checked by reading, not compiled, not run on hardware
//
#include <hip/hip_runtime.h>
#include <stdint.h>

// ---------------------------------------------------------------------------
// GCNConv for MI455X (gfx1250, wave32). Main GEMM uses v_wmma_f32_16x16x32_f16
// with a 64x128 block tile, double-buffered LDS A-staging (ds_store path),
// 4 WMMA per wave per K-chunk sharing one B fragment.
// ---------------------------------------------------------------------------

typedef __attribute__((ext_vector_type(16))) _Float16 v16h;
typedef __attribute__((ext_vector_type(8)))  _Float16 v8h;
typedef __attribute__((ext_vector_type(8)))  float    v8f;

#define KDIM 512
#define CDIM 128
#define MBLK 64            // M rows per block (4 x 16 tiles)
#define KCHUNKS (KDIM / 32)

// ---------------- utility: zero fill ----------------
__global__ void fill_zero_f32(float* __restrict__ p, long n) {
  long i = (long)blockIdx.x * blockDim.x + threadIdx.x;
  if (i < n) p[i] = 0.0f;
}

// ---------------- degree count over col = edge_index[1] ----------------
__global__ void deg_count(const int* __restrict__ ei, float* __restrict__ deg, int E) {
  int e = blockIdx.x * blockDim.x + threadIdx.x;
  if (e < E) atomicAdd(&deg[ei[E + e]], 1.0f);
}

// deg -> dinv in place (self loop adds 1, so deg >= 1 always)
__global__ void deg_to_dinv(float* __restrict__ deg, int N) {
  int i = blockIdx.x * blockDim.x + threadIdx.x;
  if (i < N) deg[i] = rsqrtf(deg[i] + 1.0f);
}

// ---------------- pack W (512x128 f32) into f16 B-fragment layout ----------
// Wp index = ((kc*8 + ntile)*32 + lane)*16 + h
// B 32x16 f16 fragment (this file's contiguous k order, same as A): n = lane&15; k = (lane<16 ? 0 : 16) + h
__global__ void pack_w(const float* __restrict__ W, _Float16* __restrict__ Wp) {
  int i = blockIdx.x * blockDim.x + threadIdx.x;   // 0 .. 65535
  if (i >= 16 * 8 * 32 * 16) return;
  int h    = i & 15;
  int lane = (i >> 4) & 31;
  int t    = (i >> 9) & 7;    // n-tile
  int kc   = i >> 12;         // k-chunk 0..15
  int n    = (lane & 15) + t * 16;
  int kb   = (lane < 16) ? 0 : 16;
  int k    = kc * 32 + kb + h;
  Wp[i] = (_Float16)W[(size_t)k * CDIM + n];
}

// ---------------- GEMM: h[M,128] = emb[M,512] @ W -------------------------
// Block: 256 threads = 8 waves; 64 M-rows x 128 N-cols. Wave w owns N-tile w
// and computes 4 M-tiles, reusing one B fragment per K-chunk across 4 WMMAs.
// A staged in LDS in fragment order (contiguous k order, same as Wp:
// m = lane&15, k = (lane<16?0:16) + h), double-buffered.
__global__ void __launch_bounds__(256, 2)
gemm_h(const float* __restrict__ emb, const _Float16* __restrict__ Wp,
       float* __restrict__ h, int Mrows) {
  __shared__ __align__(32) _Float16 AsF[2][4 * 512];   // 2 x 4KB

  const int tid  = threadIdx.x;
  const int wave = tid >> 5;            // n-tile 0..7
  const int lane = tid & 31;
  const int m0   = blockIdx.x * MBLK;

  // ---- cooperative-fill mapping: this thread produces 8 contiguous halfs
  const int fidx  = tid * 8;            // 0..2040 over [tile][lane][16]
  const int ftile = fidx >> 9;          // M-tile 0..3
  const int frem  = fidx & 511;
  const int fl    = frem >> 4;          // dest lane
  const int ffh   = frem & 15;          // dest element: 0 or 8
  const int fm    = fl & 15;
  const int fkb   = (fl < 16) ? 0 : 16;
  const int fks   = fkb + ffh;  // 8 contiguous k from here
  int frow = m0 + ftile * 16 + fm;
  if (frow > Mrows - 1) frow = Mrows - 1;   // clamp (tail block reads stay valid)
  const float* fsrc = emb + (size_t)frow * KDIM + fks;

  v8f acc[4];
  #pragma unroll
  for (int t = 0; t < 4; ++t)
    #pragma unroll
    for (int i = 0; i < 8; ++i) acc[t][i] = 0.0f;

  // ---- prologue: fill buffer 0 with K-chunk 0 (direct LDS store)
  {
    float4 e0 = *(const float4*)(fsrc + 0);
    float4 e1 = *(const float4*)(fsrc + 4);
    v8h hv;
    hv[0] = (_Float16)e0.x; hv[1] = (_Float16)e0.y;
    hv[2] = (_Float16)e0.z; hv[3] = (_Float16)e0.w;
    hv[4] = (_Float16)e1.x; hv[5] = (_Float16)e1.y;
    hv[6] = (_Float16)e1.z; hv[7] = (_Float16)e1.w;
    *(v8h*)&AsF[0][fidx] = hv;
  }
  __syncthreads();

  #pragma unroll 1
  for (int kc = 0; kc < KCHUNKS; ++kc) {
    const int cur = kc & 1;

    // B fragment for this wave's N-tile (32B per lane, L2-resident)
    v16h b = *(const v16h*)(Wp + (((size_t)(kc * 8 + wave) * 32 + lane) << 4));

    // issue next-stage global loads early
    float4 e0, e1;
    const bool more = (kc + 1 < KCHUNKS);
    if (more) {
      const float* s = fsrc + (kc + 1) * 32;
      e0 = *(const float4*)(s + 0);
      e1 = *(const float4*)(s + 4);
    }
    if (kc + 2 < KCHUNKS)
      __builtin_prefetch(fsrc + (kc + 2) * 32, 0, 3);  // global_prefetch_b8

    // load all 4 A fragments first (clause-friendly), then 4 WMMAs
    v16h a[4];
    #pragma unroll
    for (int t = 0; t < 4; ++t)
      a[t] = *(const v16h*)&AsF[cur][t * 512 + lane * 16];  // 32B, conflict-free
    #pragma unroll
    for (int t = 0; t < 4; ++t)
      acc[t] = __builtin_amdgcn_wmma_f32_16x16x32_f16(
          false, a[t], false, b, (short)0, acc[t], false, false);

    // stage next K-chunk into the other buffer (direct LDS addressing ->
    // ds_store_b128, never a flat store)
    if (more) {
      v8h hv;
      hv[0] = (_Float16)e0.x; hv[1] = (_Float16)e0.y;
      hv[2] = (_Float16)e0.z; hv[3] = (_Float16)e0.w;
      hv[4] = (_Float16)e1.x; hv[5] = (_Float16)e1.y;
      hv[6] = (_Float16)e1.z; hv[7] = (_Float16)e1.w;
      *(v8h*)&AsF[cur ^ 1][fidx] = hv;
    }
    __syncthreads();
  }

  // D 16x16 f32 layout: n = lane&15 ; vgpr r -> m = r + (lane<16 ? 0 : 8)
  const int n  = (lane & 15) + wave * 16;
  const int mb = (lane < 16) ? 0 : 8;
  if (m0 + MBLK <= Mrows) {
    // fast path: full block, unguarded stores
    float* hp = h + (size_t)(m0 + mb) * CDIM + n;
    #pragma unroll
    for (int t = 0; t < 4; ++t) {
      #pragma unroll
      for (int r = 0; r < 8; ++r)
        hp[(size_t)(t * 16 + r) * CDIM] = acc[t][r];
    }
  } else {
    // tail block only
    #pragma unroll
    for (int t = 0; t < 4; ++t) {
      const int rowbase = m0 + t * 16 + mb;
      #pragma unroll
      for (int r = 0; r < 8; ++r) {
        const int row = rowbase + r;
        if (row < Mrows) h[(size_t)row * CDIM + n] = acc[t][r];
      }
    }
  }
}

// ---------------- edge scatter: one wave per item, 4 channels per lane -----
// items 0..E-1 are edges; items E..E+N-1 are self-loops.
__global__ void scatter_edges(const int* __restrict__ ei,
                              const float* __restrict__ h,
                              const float* __restrict__ dinv,
                              float* __restrict__ outf, int E, int N) {
  long gid  = (long)blockIdx.x * blockDim.x + threadIdx.x;
  long item = gid >> 5;
  int  lane = (int)(gid & 31);
  if (item >= (long)E + N) return;

  int s, t;
  if (item < E) { s = ei[item]; t = ei[(long)E + item]; }
  else          { s = t = (int)(item - E); }

  float nrm = dinv[s] * dinv[t];
  float4 v = ((const float4*)(h + (size_t)s * CDIM))[lane];
  float* ot = outf + (size_t)t * CDIM + lane * 4;
  atomicAdd(ot + 0, v.x * nrm);
  atomicAdd(ot + 1, v.y * nrm);
  atomicAdd(ot + 2, v.z * nrm);
  atomicAdd(ot + 3, v.w * nrm);
}

// ---------------- final gather out[q] = out_full[nodes[q]] ----------------
__global__ void gather_nodes(const int* __restrict__ nodes,
                             const float* __restrict__ outf,
                             float* __restrict__ out, int Q) {
  int gid = blockIdx.x * blockDim.x + threadIdx.x;
  if (gid < Q * CDIM) {
    int q = gid >> 7, c = gid & (CDIM - 1);
    out[gid] = outf[(size_t)nodes[q] * CDIM + c];
  }
}

// ---------------------------------------------------------------------------
extern "C" void kernel_launch(void* const* d_in, const int* in_sizes, int n_in,
                              void* d_out, int out_size, void* d_ws, size_t ws_size,
                              hipStream_t stream) {
  const int*   nodes = (const int*)  d_in[0];   // [10000]
  const int*   ei    = (const int*)  d_in[1];   // [2, E] row-major
  const float* emb   = (const float*)d_in[2];   // [N, 512]
  const float* W     = (const float*)d_in[3];   // [512, 128]

  const int Q = in_sizes[0];
  const int E = in_sizes[1] / 2;
  const int N = in_sizes[2] / KDIM;

  // workspace carve-up
  char* ws = (char*)d_ws;
  float* h    = (float*)ws;  ws += (size_t)N * CDIM * sizeof(float);
  float* outf = (float*)ws;  ws += (size_t)N * CDIM * sizeof(float);
  float* dinv = (float*)ws;  ws += (size_t)N * sizeof(float);
  ws = (char*)(((uintptr_t)ws + 31) & ~(uintptr_t)31);
  _Float16* Wp = (_Float16*)ws;                 // 16*8*32*16 halfs = 128 KB

  const int T = 256;

  // 1. pack W into f16 fragment layout (L2-resident, reused by all blocks)
  pack_w<<<(16 * 8 * 32 * 16 + T - 1) / T, T, 0, stream>>>(W, Wp);

  // 2. zero degree accumulator and output accumulator
  fill_zero_f32<<<(N + T - 1) / T, T, 0, stream>>>(dinv, N);
  long on = (long)N * CDIM;
  fill_zero_f32<<<(int)((on + T - 1) / T), T, 0, stream>>>(outf, on);

  // 3. symmetric-norm degrees: count into col, then rsqrt(deg+1) in place
  deg_count<<<(E + T - 1) / T, T, 0, stream>>>(ei, dinv, E);
  deg_to_dinv<<<(N + T - 1) / T, T, 0, stream>>>(dinv, N);

  // 4. main GEMM h = emb @ W via WMMA, 64-row block tiles
  gemm_h<<<(N + MBLK - 1) / MBLK, T, 0, stream>>>(emb, Wp, h, N);

  // 5. gather + scale + scatter-add over edges and self-loops
  long items = ((long)E + N) * 32;
  scatter_edges<<<(int)((items + T - 1) / T), T, 0, stream>>>(ei, h, dinv, outf, E, N);

  // 6. final gather of queried nodes
  gather_nodes<<<(Q * CDIM + T - 1) / T, T, 0, stream>>>(nodes, outf, (float*)d_out, Q);
}
